// Decoder_output_as_input_54434415509952
// MI455X (gfx1250) — matrix operand order repaired and checked by reading, not compiled, not run on hardware
//
#include <hip/hip_runtime.h>
#include <cstdint>

typedef __attribute__((ext_vector_type(16))) __bf16 v16bf;
typedef __attribute__((ext_vector_type(8)))  float  v8f;

#define HID 512
#define PH  32
#define NKT 16      // 512/32 K-tiles
#define WG_ROWS 16  // batch rows per workgroup

// ---- helpers ---------------------------------------------------------------
__device__ __forceinline__ unsigned short f2bf(float f) {
  // round-to-nearest-even fp32 -> bf16 (finite inputs)
  unsigned int x = __builtin_bit_cast(unsigned int, f);
  unsigned int r = (x + 0x7FFFu + ((x >> 16) & 1u)) >> 16;
  return (unsigned short)r;
}
__device__ __forceinline__ float bf2f(unsigned short u) {
  unsigned int x = ((unsigned int)u) << 16;
  return __builtin_bit_cast(float, x);
}
__device__ __forceinline__ float fast_sigmoid(float x) {
  return 1.0f / (1.0f + __expf(-x));
}
__device__ __forceinline__ float fast_tanh(float x) {
  return 1.0f - 2.0f / (__expf(2.0f * x) + 1.0f);
}

#define WMMA_BF16(A, B, C) \
  __builtin_amdgcn_wmma_f32_16x16x32_bf16(false, (A), false, (B), (short)0, (C), false, false)

// ---- prep: swizzle weights into per-lane WMMA B-fragment order -------------
// B-frag element o = ((nt*16 + kt)*32 + lane)*16 + e  holds  W[n][k] with
//   n = nt*16 + (lane&15),  k = kt*32 + (lane>>4)*8 + (e&7) + ((e&8) ? 16 : 0)
__global__ void prep_weights(const float* __restrict__ Wih,
                             const float* __restrict__ Whh,
                             const float* __restrict__ bih,
                             const float* __restrict__ bhh,
                             unsigned short* __restrict__ wih_sw,
                             unsigned short* __restrict__ wsum_sw,
                             float* __restrict__ bsum) {
  int o = blockIdx.x * 256 + threadIdx.x;
  const int TOT = 2048 * 512;
  if (o < TOT) {
    int e    = o & 15;
    int lane = (o >> 4) & 31;
    int kt   = (o >> 9) & 15;
    int nt   = o >> 13;
    int n = nt * 16 + (lane & 15);
    int k = kt * 32 + (lane >> 4) * 8 + (e & 7) + ((e & 8) ? 16 : 0);
    float a = Wih[n * 512 + k];
    float b = Whh[n * 512 + k];
    wih_sw[o]  = f2bf(a);
    wsum_sw[o] = f2bf(a + b);
  }
  if (o < 2048) bsum[o] = bih[o] + bhh[o];
}

// ---- main LSTM decoder -----------------------------------------------------
__global__ __launch_bounds__(256)
void lstm_decoder(const float* __restrict__ z,
                  const unsigned short* __restrict__ wih_sw,
                  const unsigned short* __restrict__ wsum_sw,
                  const float* __restrict__ bsum_g,
                  const float* __restrict__ Wd,
                  const float* __restrict__ bd,
                  float* __restrict__ y) {
  // h buffers in A-fragment swizzled bf16: [kt(16)][lane(32)][e(16)]
  __shared__ __attribute__((aligned(32))) unsigned short hbuf[2][NKT * 32 * 16];
  __shared__ float bsum_s[4 * HID];
  __shared__ float wd_s[2 * HID];
  __shared__ float ypart[256];

  const int tid  = threadIdx.x;
  const int lane = tid & 31;
  const int wv   = tid >> 5;              // wave id 0..7
  const int b0   = blockIdx.x * WG_ROWS;  // first batch row of this WG

  // stage biases and dense weights
  for (int i = tid; i < 4 * HID; i += 256) bsum_s[i] = bsum_g[i];
  for (int i = tid; i < 2 * HID; i += 256) wd_s[i] = Wd[i];

  // init h buffer 0 from z (x0 = z), swizzled bf16
  for (int j = 0; j < 32; ++j) {
    int idx = tid + 256 * j;           // 0..8191 over 16x512
    int M = idx >> 9, k = idx & 511;
    float v = z[(size_t)(b0 + M) * HID + k];
    int kt = k >> 5, ko = k & 31;
    int lp = M + ((ko & 8) ? 16 : 0);
    int e  = (ko & 7) + ((ko & 16) ? 8 : 0);
    hbuf[0][(kt * 32 + lp) * 16 + e] = f2bf(v);
  }
  __syncthreads();

  const v8f zero8 = {0.f, 0.f, 0.f, 0.f, 0.f, 0.f, 0.f, 0.f};
  v8f cst[4] = {zero8, zero8, zero8, zero8};   // c-state, 4 hidden tiles/wave

  int p = 0;
  for (int t = 0; t < PH; ++t) {
    const unsigned short* Bw = (t == 0) ? wih_sw : wsum_sw;
    const unsigned short* cur = hbuf[p];
    unsigned short* nxt = hbuf[1 - p];

    // ---- GEMM phase: gates = h @ W^T, 16 accumulators (4 hidden tiles x 4 gates)
    v8f acc[16];
    #pragma unroll
    for (int i = 0; i < 16; ++i) acc[i] = zero8;

    // element offsets (ushort units): kt: 512, hidx: 8*16*512=65536, gate: 32*16*512=262144
    const unsigned short* bbase = Bw + (size_t)(wv * 16) * 512 + lane * 16;

    #pragma unroll 1
    for (int kt = 0; kt < NKT; ++kt) {
      v16bf a = *(const v16bf*)(cur + (kt * 32 + lane) * 16);
      const unsigned short* bk = bbase + (size_t)kt * 512;
      #pragma unroll
      for (int hidx = 0; hidx < 4; ++hidx) {
        const unsigned short* bh = bk + (size_t)hidx * 65536;
        v16bf b0v = *(const v16bf*)(bh);
        v16bf b1v = *(const v16bf*)(bh + 262144);
        v16bf b2v = *(const v16bf*)(bh + 524288);
        v16bf b3v = *(const v16bf*)(bh + 786432);
        acc[hidx * 4 + 0] = WMMA_BF16(a, b0v, acc[hidx * 4 + 0]);
        acc[hidx * 4 + 1] = WMMA_BF16(a, b1v, acc[hidx * 4 + 1]);
        acc[hidx * 4 + 2] = WMMA_BF16(a, b2v, acc[hidx * 4 + 2]);
        acc[hidx * 4 + 3] = WMMA_BF16(a, b3v, acc[hidx * 4 + 3]);
      }
    }

    // ---- elementwise LSTM combine; C/D layout: lane N=lane&15, VGPR r: M=r+8*(lane>>4)
    #pragma unroll
    for (int hidx = 0; hidx < 4; ++hidx) {
      const int hj  = wv + 8 * hidx;
      const int c   = lane & 15;
      const int nc  = hj * 16 + c;
      const float bi  = bsum_s[nc];
      const float bf_ = bsum_s[512 + nc];
      const float bg  = bsum_s[1024 + nc];
      const float bo  = bsum_s[1536 + nc];
      const int kt_w = hj >> 1;
      const int ko   = ((hj & 1) << 4) | c;
      const int lp0  = 8 * (lane >> 4) + ((ko & 8) ? 16 : 0);
      const int e    = (ko & 7) + ((ko & 16) ? 8 : 0);
      #pragma unroll
      for (int r = 0; r < 8; ++r) {
        float iv = fast_sigmoid(acc[hidx * 4 + 0][r] + bi);
        float fv = fast_sigmoid(acc[hidx * 4 + 1][r] + bf_);
        float gv = fast_tanh(acc[hidx * 4 + 2][r] + bg);
        float ov = fast_sigmoid(acc[hidx * 4 + 3][r] + bo);
        float cn = fv * cst[hidx][r] + iv * gv;
        float hv = ov * fast_tanh(cn);
        cst[hidx][r] = cn;
        nxt[(kt_w * 32 + (r + lp0)) * 16 + e] = f2bf(hv);
      }
    }
    __syncthreads();   // all h_new written; all reads of `cur` done

    // ---- dense head: y[b, t, :] = h_new @ Wd^T + bd  (OUT=2)
    {
      int row = tid >> 4;
      int pr  = tid & 15;
      int o   = pr & 1;
      int kc  = pr >> 1;          // 8 chunks of 64
      float s = 0.f;
      #pragma unroll 8
      for (int k = kc * 64; k < kc * 64 + 64; ++k) {
        int kt = k >> 5, ko2 = k & 31;
        int lp = row + ((ko2 & 8) ? 16 : 0);
        int e2 = (ko2 & 7) + ((ko2 & 16) ? 8 : 0);
        s += bf2f(nxt[(kt * 32 + lp) * 16 + e2]) * wd_s[o * HID + k];
      }
      ypart[tid] = s;
    }
    __syncthreads();
    if (tid < 32) {
      int row = tid >> 1, o = tid & 1;
      float s = bd[o];
      #pragma unroll
      for (int kc = 0; kc < 8; ++kc) s += ypart[row * 16 + (o + 2 * kc)];
      y[(size_t)(b0 + row) * (PH * 2) + t * 2 + o] = s;
    }
    p ^= 1;
  }
}

// ---- launch ----------------------------------------------------------------
extern "C" void kernel_launch(void* const* d_in, const int* in_sizes, int n_in,
                              void* d_out, int out_size, void* d_ws, size_t ws_size,
                              hipStream_t stream) {
  (void)in_sizes; (void)n_in; (void)out_size; (void)ws_size;
  // inputs: 0=hist(unused) 1=z 2=W_ih 3=W_hh 4=b_ih 5=b_hh 6=W_d 7=b_d
  const float* z   = (const float*)d_in[1];
  const float* Wih = (const float*)d_in[2];
  const float* Whh = (const float*)d_in[3];
  const float* bih = (const float*)d_in[4];
  const float* bhh = (const float*)d_in[5];
  const float* Wd  = (const float*)d_in[6];
  const float* bd  = (const float*)d_in[7];
  float* y = (float*)d_out;

  // workspace: [0,2MB) W_sum swizzled bf16, [2MB,4MB) W_ih swizzled bf16,
  //            [4MB,+8KB) b_sum fp32
  unsigned short* wsum_sw = (unsigned short*)d_ws;
  unsigned short* wih_sw  = (unsigned short*)((char*)d_ws + 2u * 1024 * 1024);
  float*          bsum    = (float*)((char*)d_ws + 4u * 1024 * 1024);

  prep_weights<<<(2048 * 512) / 256, 256, 0, stream>>>(Wih, Whh, bih, bhh,
                                                       wih_sw, wsum_sw, bsum);
  lstm_decoder<<<2048 / WG_ROWS, 256, 0, stream>>>(z, wih_sw, wsum_sw, bsum,
                                                   Wd, bd, y);
}
